// WaveletBlock_46806553592454
// MI455X (gfx1250) — matrix operand order repaired and checked by reading, not compiled, not run on hardware
//
#include <hip/hip_runtime.h>
#include <math.h>

typedef __attribute__((ext_vector_type(16))) _Float16 v16h;
typedef __attribute__((ext_vector_type(8)))  float    v8f;

union FragAB {
    v16h v;
    _Float16 h[16];
    uint4 q[2];
};

// ---------------------------------------------------------------------------
// Weight conversion: OIHW fp32 -> per-lane-swizzled f16 B-fragments.
// Fragment layout: [nt][kb][lane 0..31][16 halfs] (32B contiguous per lane).
// lane<16 holds K = kb*32 + 0..15 for column n = nt*16 + lane;
// lane>=16 holds K = kb*32 + 16..31 for column n = nt*16 + (lane-16).
// k_gemm = (r*S + s)*Cin + c  (channel fastest => contiguous NHWC A loads).
// ---------------------------------------------------------------------------
__global__ void wfrag_kernel(const float* __restrict__ w, _Float16* __restrict__ frag,
                             int Kout, int Cin, int R, int S)
{
    const int Kt = Cin * R * S;
    const long long total = (long long)Kt * Kout;
    long long i = (long long)blockIdx.x * blockDim.x + threadIdx.x;
    if (i >= total) return;
    int j = (int)(i & 15);
    long long t = i >> 4;
    int lane = (int)(t & 31); t >>= 5;
    const int KB = Kt / 32;
    int kb = (int)(t % KB);
    int nt = (int)(t / KB);
    int k = kb * 32 + ((lane >> 4) << 4) + j;
    int n = nt * 16 + (lane & 15);
    int c = k % Cin;
    int rs = k / Cin;
    int r = rs / S, s = rs % S;
    float val = w[(((long long)n * Cin + c) * R + r) * S + s];
    frag[i] = (_Float16)val;
}

// ---------------------------------------------------------------------------
// Implicit-GEMM conv, stride 1, same-size output. All shape params are
// compile-time so every div/mod folds to shifts and the k-loop is pure
// load/load/wmma. Each wave computes a 32(M) x 64(N) macro-tile:
// 2 M-tiles x 4 N-tiles = 8 v_wmma per 32-channel block, A reused 4x, B 2x.
// EPI: 0 = store f16 NHWC (optional relu)
//      1 = store f32 NHWC
//      2 = store f32 NHWC + residual add (resid, same layout)
//      3 = store f32 NCHW (plain)
//      4 = d_out NCHW: out = relu(acc) + out   (accumulate onto existing)
// ---------------------------------------------------------------------------
template <int R, int S, int PAD, int CIN, int KOUT, int H, int W, int EPI, bool RELU>
__global__ __launch_bounds__(128) void conv_wmma_kernel(
    const _Float16* __restrict__ A,     // NHWC f16 (NB,H,W,CIN)
    const _Float16* __restrict__ Bfrag, // swizzled weights
    float* __restrict__ outF,
    _Float16* __restrict__ outH,
    const float* __restrict__ resid)
{
    constexpr int KB = CIN * R * S / 32;  // total 32-wide k-blocks
    constexpr int CB = CIN / 32;          // k-blocks per (r,s) tap
    const int tid  = threadIdx.x;
    const int lane = tid & 31;
    const int wv   = tid >> 5;
    const int half = lane >> 4;
    const int lr   = lane & 15;
    const int ngrp = blockIdx.y;                           // 64-channel group
    const long long mt = (long long)blockIdx.x * 4 + wv;   // 32-row super-tile

    // Row decomposition for the two 16-row tiles (W,H are powers of two).
    int pp[2], qq[2], nn[2];
#pragma unroll
    for (int t = 0; t < 2; ++t) {
        const long long row = mt * 32 + t * 16 + lr;
        qq[t] = (int)(row % W);
        const long long u = row / W;
        pp[t] = (int)(u % H);
        nn[t] = (int)(u / H);
    }

    const _Float16* bbase = Bfrag + (((long long)(ngrp * 4) * KB) * 32 + lane) * 16;

    v8f acc[2][4] = {};

#pragma unroll
    for (int r = 0; r < R; ++r) {
#pragma unroll
        for (int s = 0; s < S; ++s) {
            const int kb0 = (r * S + s) * CB;
            const _Float16* ab[2];
            bool ok[2];
#pragma unroll
            for (int t = 0; t < 2; ++t) {
                const int y = pp[t] + r - PAD;
                const int x = qq[t] + s - PAD;
                ok[t] = (unsigned)y < (unsigned)H && (unsigned)x < (unsigned)W;
                ab[t] = A + (((long long)nn[t] * H + y) * W + x) * CIN + half * 16;
            }
#pragma unroll
            for (int cb = 0; cb < CB; ++cb) {
                FragAB af[2];
#pragma unroll
                for (int t = 0; t < 2; ++t) {
                    if (ok[t]) {
                        af[t].q[0] = *reinterpret_cast<const uint4*>(ab[t] + cb * 32);
                        af[t].q[1] = *reinterpret_cast<const uint4*>(ab[t] + cb * 32 + 8);
                    } else {
                        af[t].q[0] = make_uint4(0, 0, 0, 0);
                        af[t].q[1] = make_uint4(0, 0, 0, 0);
                    }
                }
                if (cb + 1 < CB)   // weight-stream prefetch -> global_prefetch_b8
                    __builtin_prefetch(bbase + (long long)(kb0 + cb + 1) * (32 * 16), 0, 1);
#pragma unroll
                for (int j = 0; j < 4; ++j) {
                    FragAB bf;
                    const _Float16* bp =
                        bbase + ((long long)j * KB + kb0 + cb) * (32 * 16);
                    bf.q[0] = *reinterpret_cast<const uint4*>(bp);
                    bf.q[1] = *reinterpret_cast<const uint4*>(bp + 8);
#pragma unroll
                    for (int t = 0; t < 2; ++t)
                        acc[t][j] = __builtin_amdgcn_wmma_f32_16x16x32_f16(
                            false, af[t].v, false, bf.v, (short)0, acc[t][j],
                            false, false);
                }
            }
        }
    }

#pragma unroll
    for (int t = 0; t < 2; ++t) {
#pragma unroll
        for (int j = 0; j < 4; ++j) {
            const int col = (ngrp * 4 + j) * 16 + lr;
#pragma unroll
            for (int v = 0; v < 8; ++v) {
                const long long row = mt * 32 + t * 16 + v + half * 8;
                float val = acc[t][j][v];
                if (RELU) val = fmaxf(val, 0.f);
                if (EPI == 0) {
                    outH[row * KOUT + col] = (_Float16)val;
                } else if (EPI == 1) {
                    outF[row * KOUT + col] = val;
                } else if (EPI == 2) {
                    outF[row * KOUT + col] = val + resid[row * KOUT + col];
                } else {
                    const int q2 = (int)(row % W);
                    const long long u = row / W;
                    const int p2 = (int)(u % H);
                    const int n2 = (int)(u / H);
                    const long long oi = (((long long)n2 * KOUT + col) * H + p2) * W + q2;
                    if (EPI == 3) outF[oi] = val;
                    else          outF[oi] = fmaxf(val, 0.f) + outF[oi];
                }
            }
        }
    }
}

// ---------------------------------------------------------------------------
// bwt: x NCHW f32 (NB,C,2H2,2W2) -> x_bwt NHWC (NB,H2,W2,8C) fp32 + f16
// ---------------------------------------------------------------------------
__global__ void bwt_kernel(const float* __restrict__ x, float* __restrict__ obF,
                           _Float16* __restrict__ obH, int NB, int C, int H2, int W2)
{
    const long long total = (long long)NB * C * H2 * W2;
    long long i = (long long)blockIdx.x * blockDim.x + threadIdx.x;
    if (i >= total) return;
    const int w2 = (int)(i % W2);
    long long t = i / W2;
    const int h2 = (int)(t % H2); t /= H2;
    const int c = (int)(t % C);
    const int n = (int)(t / C);
    const int H = H2 * 2, W = W2 * 2;
    const float* base = x + (((long long)n * C + c) * H + 2 * h2) * W + 2 * w2;
    const float x1 = base[0]     * 0.5f;  // even row, even col
    const float x3 = base[1]     * 0.5f;  // even row, odd col
    const float x2 = base[W]     * 0.5f;  // odd row, even col
    const float x4 = base[W + 1] * 0.5f;  // odd row, odd col
    float b[8];
    b[0] =  x1 + x2 + x3 + x4;
    b[1] = -x1 - x2 + x3 + x4;
    b[2] = -x1 + x2 - x3 + x4;
    b[3] =  x1 - x2 - x3 + x4;
    b[4] =  x1 + x2 - x3 - x4;
    b[5] = -x1 + x2 + x3 - x4;
    b[6] =  x1 - x2 + x3 - x4;
    b[7] = -x1 - x2 - x3 - x4;
    const long long ob = (((long long)n * H2 + h2) * W2 + w2) * (8 * C) + c;
#pragma unroll
    for (int k = 0; k < 8; ++k) {
        obF[ob + (long long)k * C] = b[k];
        obH[ob + (long long)k * C] = (_Float16)b[k];
    }
}

// x NCHW f32 -> NHWC f16
__global__ void x_nhwc_kernel(const float* __restrict__ x, _Float16* __restrict__ o,
                              int NB, int C, int H, int W)
{
    const long long total = (long long)NB * C * H * W;
    long long i = (long long)blockIdx.x * blockDim.x + threadIdx.x;
    if (i >= total) return;
    const int c = (int)(i % C);
    long long t = i / C;
    const int w = (int)(t % W); t /= W;
    const int h = (int)(t % H);
    const int n = (int)(t / H);
    o[i] = (_Float16)x[(((long long)n * C + c) * H + h) * W + w];
}

// channel max/mean per pixel over NHWC res2 -> pooled[pix*2 + {0:max,1:avg}]
__global__ void pool_kernel(const float* __restrict__ res2, float* __restrict__ pooled,
                            long long NPix, int C)
{
    long long i = (long long)blockIdx.x * blockDim.x + threadIdx.x;
    if (i >= NPix) return;
    const float* p = res2 + i * C;
    float mx = -3.402823466e38f, sm = 0.f;
    for (int c = 0; c < C; ++c) { float v = p[c]; mx = fmaxf(mx, v); sm += v; }
    pooled[i * 2 + 0] = mx;
    pooled[i * 2 + 1] = sm / (float)C;
}

// 5x5 conv (2ch -> 1ch, pad 2) + sigmoid -> y map
__global__ void saly_kernel(const float* __restrict__ pooled, const float* __restrict__ wsal,
                            float* __restrict__ yout, int NB, int H, int W)
{
    const long long total = (long long)NB * H * W;
    long long i = (long long)blockIdx.x * blockDim.x + threadIdx.x;
    if (i >= total) return;
    const int q = (int)(i % W);
    long long t = i / W;
    const int p = (int)(t % H);
    const int n = (int)(t / H);
    float acc = 0.f;
    for (int ch = 0; ch < 2; ++ch)
        for (int dy = 0; dy < 5; ++dy)
            for (int dx = 0; dx < 5; ++dx) {
                const int yy = p + dy - 2, xx = q + dx - 2;
                if ((unsigned)yy < (unsigned)H && (unsigned)xx < (unsigned)W)
                    acc += pooled[(((long long)n * H + yy) * W + xx) * 2 + ch] *
                           wsal[(ch * 5 + dy) * 5 + dx];
            }
    yout[i] = 1.f / (1.f + expf(-acc));
}

// weighted spatial mean per (n,c): mean over pixels of res2*y  (LDS reduce)
__global__ void calmean_kernel(const float* __restrict__ res2, const float* __restrict__ y,
                               float* __restrict__ calin, int C, int HW)
{
    const int c = blockIdx.x % C;
    const int n = blockIdx.x / C;
    __shared__ float red[256];
    float s = 0.f;
    for (int i = threadIdx.x; i < HW; i += 256)
        s += res2[((long long)n * HW + i) * C + c] * y[(long long)n * HW + i];
    red[threadIdx.x] = s;
    __syncthreads();
    for (int st = 128; st > 0; st >>= 1) {
        if (threadIdx.x < st) red[threadIdx.x] += red[threadIdx.x + st];
        __syncthreads();
    }
    if (threadIdx.x == 0) calin[blockIdx.x] = red[0] / (float)HW;
}

__global__ void ca_hidden_kernel(const float* __restrict__ calin, const float* __restrict__ w1,
                                 float* __restrict__ hidden, int C, int Hd, int NB)
{
    const int i = blockIdx.x * blockDim.x + threadIdx.x;
    if (i >= NB * Hd) return;
    const int k = i % Hd, n = i / Hd;
    float s = 0.f;
    for (int c = 0; c < C; ++c) s += w1[(long long)k * C + c] * calin[n * C + c];
    hidden[i] = fmaxf(s, 0.f);
}

__global__ void ca_out_kernel(const float* __restrict__ hidden, const float* __restrict__ w2,
                              float* __restrict__ ca, int Hd, int C, int NB)
{
    const int i = blockIdx.x * blockDim.x + threadIdx.x;
    if (i >= NB * C) return;
    const int c = i % C, n = i / C;
    float s = 0.f;
    for (int j = 0; j < Hd; ++j) s += w2[(long long)c * Hd + j] * hidden[n * Hd + j];
    ca[i] = 1.f / (1.f + expf(-s));
}

// res2 * y * ca -> f16 NHWC
__global__ void scale_kernel(const float* __restrict__ res2, const float* __restrict__ y,
                             const float* __restrict__ ca, _Float16* __restrict__ out,
                             int C, int HW, long long total)
{
    long long i = (long long)blockIdx.x * blockDim.x + threadIdx.x;
    if (i >= total) return;
    const int c = (int)(i % C);
    const long long pix = i / C;
    const int n = (int)(pix / HW);
    out[i] = (_Float16)(res2[i] * y[pix] * ca[(long long)n * C + c]);
}

// ibwt: res3 NHWC f32 (NB,H2,W2,8C) -> wavelet NHWC f16 (NB,2H2,2W2,C)
__global__ void ibwt_kernel(const float* __restrict__ res3, _Float16* __restrict__ wav,
                            int NB, int C, int H2, int W2)
{
    const long long total = (long long)NB * H2 * W2 * C;
    long long i = (long long)blockIdx.x * blockDim.x + threadIdx.x;
    if (i >= total) return;
    const int c = (int)(i % C);
    long long t = i / C;
    const int w2 = (int)(t % W2); t /= W2;
    const int h2 = (int)(t % H2);
    const int n = (int)(t / H2);
    const long long base = (((long long)n * H2 + h2) * W2 + w2) * (8 * C) + c;
    float xs[8];
#pragma unroll
    for (int k = 0; k < 8; ++k) xs[k] = res3[base + (long long)k * C] * 0.5f;
    const float a00 = xs[0] - xs[1] - xs[2] + xs[3] + xs[4] - xs[5] + xs[6] - xs[7];
    const float a10 = xs[0] - xs[1] + xs[2] - xs[3] - xs[4] + xs[5] - xs[6] + xs[7];
    const float a01 = xs[0] + xs[1] - xs[2] - xs[3] - xs[4] - xs[5] + xs[6] + xs[7];
    const float a11 = xs[0] + xs[1] + xs[2] + xs[3] + xs[4] + xs[5] + xs[6] + xs[7];
    const int H = 2 * H2, W = 2 * W2;
    const long long o00 = (((long long)n * H + 2 * h2) * W + 2 * w2) * C + c;
    wav[o00]                        = (_Float16)a00;
    wav[o00 + C]                    = (_Float16)a01;
    wav[o00 + (long long)W * C]     = (_Float16)a10;
    wav[o00 + (long long)W * C + C] = (_Float16)a11;
}

// ---------------------------------------------------------------------------
extern "C" void kernel_launch(void* const* d_in, const int* in_sizes, int n_in,
                              void* d_out, int out_size, void* d_ws, size_t ws_size,
                              hipStream_t stream)
{
    (void)in_sizes; (void)n_in; (void)out_size; (void)ws_size;
    const float* x       = (const float*)d_in[0];
    const float* w_body1 = (const float*)d_in[1];
    const float* w_body2 = (const float*)d_in[2];
    const float* w_sal   = (const float*)d_in[3];
    const float* w_ca1   = (const float*)d_in[4];
    const float* w_ca2   = (const float*)d_in[5];
    const float* w_1x1   = (const float*)d_in[6];
    const float* w_3x3   = (const float*)d_in[7];
    const float* w_final = (const float*)d_in[8];
    float* out = (float*)d_out;

    const int NB = 4, C = 64, H = 256, W = 256, H2 = 128, W2 = 128, C8 = 512, Hd = 32;
    const long long PIX2 = (long long)NB * H2 * W2;   // 65536
    const long long PIX  = (long long)NB * H * W;     // 262144

    char* ws = (char*)d_ws;
    auto alloc = [&](size_t bytes) {
        void* p = (void*)ws;
        ws += (bytes + 255) & ~(size_t)255;
        return p;
    };

    float*    xbwtF  = (float*)   alloc(PIX2 * C8 * sizeof(float));
    _Float16* xbwtH  = (_Float16*)alloc(PIX2 * C8 * sizeof(_Float16));
    _Float16* xH     = (_Float16*)alloc(PIX  * C  * sizeof(_Float16));
    _Float16* wf1    = (_Float16*)alloc((size_t)4608 * 64  * sizeof(_Float16));
    _Float16* wf2    = (_Float16*)alloc((size_t)576  * 512 * sizeof(_Float16));
    _Float16* wf1x1  = (_Float16*)alloc((size_t)512  * 512 * sizeof(_Float16));
    _Float16* wf3    = (_Float16*)alloc((size_t)576  * 64  * sizeof(_Float16));
    _Float16* wfF    = (_Float16*)alloc((size_t)64   * 64  * sizeof(_Float16));
    _Float16* res1H  = (_Float16*)alloc(PIX2 * C  * sizeof(_Float16));
    float*    res2F  = (float*)   alloc(PIX2 * C8 * sizeof(float));
    float*    pooled = (float*)   alloc(PIX2 * 2  * sizeof(float));
    float*    ymap   = (float*)   alloc(PIX2 * sizeof(float));
    float*    calin  = (float*)   alloc((size_t)NB * C8 * sizeof(float));
    float*    hidden = (float*)   alloc((size_t)NB * Hd * sizeof(float));
    float*    cav    = (float*)   alloc((size_t)NB * C8 * sizeof(float));
    _Float16* salcal = (_Float16*)alloc(PIX2 * C8 * sizeof(_Float16));
    float*    res3F  = (float*)   alloc(PIX2 * C8 * sizeof(float));
    _Float16* wavH   = (_Float16*)alloc(PIX  * C  * sizeof(_Float16));

    auto blocks = [](long long tot, int bs) { return (unsigned)((tot + bs - 1) / bs); };

    // 1. weight fragments
    wfrag_kernel<<<blocks((long long)4608 * 64, 256), 256, 0, stream>>>(w_body1, wf1,  64,  512, 3, 3);
    wfrag_kernel<<<blocks((long long)576 * 512, 256), 256, 0, stream>>>(w_body2, wf2,  512, 64,  3, 3);
    wfrag_kernel<<<blocks((long long)512 * 512, 256), 256, 0, stream>>>(w_1x1,   wf1x1,512, 512, 1, 1);
    wfrag_kernel<<<blocks((long long)576 * 64,  256), 256, 0, stream>>>(w_3x3,   wf3,  64,  64,  3, 3);
    wfrag_kernel<<<blocks((long long)64 * 64,   256), 256, 0, stream>>>(w_final, wfF,  64,  64,  1, 1);

    // 2. bwt + x -> NHWC f16
    bwt_kernel<<<blocks((long long)NB * C * H2 * W2, 256), 256, 0, stream>>>(x, xbwtF, xbwtH, NB, C, H2, W2);
    x_nhwc_kernel<<<blocks(PIX * C, 256), 256, 0, stream>>>(x, xH, NB, C, H, W);

    // 3. body1: 3x3, 512->64, relu, f16 out   (M=65536 -> 512 blocks)
    conv_wmma_kernel<3, 3, 1, 512, 64, 128, 128, 0, true>
        <<<dim3(512, 1), 128, 0, stream>>>(xbwtH, wf1, nullptr, res1H, nullptr);

    // 4. body2: 3x3, 64->512, f32 out
    conv_wmma_kernel<3, 3, 1, 64, 512, 128, 128, 1, false>
        <<<dim3(512, 8), 128, 0, stream>>>(res1H, wf2, res2F, nullptr, nullptr);

    // 5. spatial attention
    pool_kernel<<<blocks(PIX2, 256), 256, 0, stream>>>(res2F, pooled, PIX2, C8);
    saly_kernel<<<blocks(PIX2, 256), 256, 0, stream>>>(pooled, w_sal, ymap, NB, H2, W2);

    // 6. channel attention on sal output
    calmean_kernel<<<NB * C8, 256, 0, stream>>>(res2F, ymap, calin, C8, H2 * W2);
    ca_hidden_kernel<<<1, 128, 0, stream>>>(calin, w_ca1, hidden, C8, Hd, NB);
    ca_out_kernel<<<blocks((long long)NB * C8, 256), 256, 0, stream>>>(hidden, w_ca2, cav, Hd, C8, NB);
    scale_kernel<<<blocks(PIX2 * C8, 256), 256, 0, stream>>>(res2F, ymap, cav, salcal, C8, H2 * W2, PIX2 * C8);

    // 7. 1x1 512->512 + x_bwt residual, f32 out
    conv_wmma_kernel<1, 1, 0, 512, 512, 128, 128, 2, false>
        <<<dim3(512, 8), 128, 0, stream>>>(salcal, wf1x1, res3F, nullptr, xbwtF);

    // 8. ibwt -> wavelet path f16
    ibwt_kernel<<<blocks((long long)NB * H2 * W2 * C, 256), 256, 0, stream>>>(res3F, wavH, NB, C, H2, W2);

    // 9. final 1x1 on original x -> d_out NCHW (plain store, initializes d_out)
    conv_wmma_kernel<1, 1, 0, 64, 64, 256, 256, 3, false>
        <<<dim3(2048, 1), 128, 0, stream>>>(xH, wfF, out, nullptr, nullptr);

    // 10. 3x3 64->64 on wavelet path: d_out += relu(acc)
    conv_wmma_kernel<3, 3, 1, 64, 64, 256, 256, 4, false>
        <<<dim3(2048, 1), 128, 0, stream>>>(wavH, wf3, out, nullptr, nullptr);
}
